// SelfAttention_75067438400000
// MI455X (gfx1250) — matrix operand order repaired and checked by reading, not compiled, not run on hardware
//
#include <hip/hip_runtime.h>
#include <hip/hip_bf16.h>
#include <stdint.h>
#include <stddef.h>

// ---------------------------------------------------------------------------
// Fused self-attention (QKV GEMM + RoPE, flash attention, out projection)
// for MI455X / gfx1250: wave32, V_WMMA_F32_16X16X32_BF16, async global->LDS
// staging of shared GEMM A-tiles (ASYNCcnt path) when the toolchain has it.
// ---------------------------------------------------------------------------

typedef __attribute__((ext_vector_type(16))) __bf16 v16bf;
typedef __attribute__((ext_vector_type(8)))  __bf16 v8bf;
typedef __attribute__((ext_vector_type(8)))  float  v8f;
typedef __attribute__((ext_vector_type(4)))  int    v4i;

#define B_    4
#define L_    2048
#define D_    1024
#define H_    16
#define DH_   64
#define M_    (B_ * L_)     // 8192 rows
#define NQKV_ 3072

// ---- gfx1250 async global->LDS support (guarded) ---------------------------
#if defined(__has_builtin)
# if __has_builtin(__builtin_amdgcn_global_load_async_to_lds_b128)
#  define HAVE_ASYNC_LDS 1
# endif
#endif
#ifndef HAVE_ASYNC_LDS
# define HAVE_ASYNC_LDS 0
#endif

#if HAVE_ASYNC_LDS
// clang prints addrspace(1) as "__device__" and addrspace(3) as "__shared__";
// the builtin expects int4 pointers in those address spaces.
typedef __attribute__((address_space(1))) v4i gv4i_t;
typedef __attribute__((address_space(3))) v4i lv4i_t;
#endif

__device__ __forceinline__ void wait_async_done() {
#if HAVE_ASYNC_LDS
# if __has_builtin(__builtin_amdgcn_s_wait_asynccnt)
  __builtin_amdgcn_s_wait_asynccnt(0);
# else
  asm volatile("s_wait_asynccnt 0x0" ::: "memory");
# endif
#endif
}

// ---- WMMA wrapper ----------------------------------------------------------
__device__ __forceinline__ v8f wmma_bf16(v16bf a, v16bf b, v8f c) {
  // (neg_a, A, neg_b, B, c_mod, C, reuse_a, reuse_b)
  return __builtin_amdgcn_wmma_f32_16x16x32_bf16(false, a, false, b,
                                                 (short)0, c, false, false);
}

// ---- fragment loaders ------------------------------------------------------
// A (16x32 bf16, row-major source, row stride lda elements):
//   lanes 0-15 : row = lane,   K {k0+0..15}
//   lanes 16-31: row = lane-16, K {k0+16..31}
__device__ __forceinline__ v16bf load_a_frag(const __bf16* A, int lda,
                                             int row_base, int k0, int lane) {
  const int half = lane >> 4;
  const int row  = lane & 15;
  const __bf16* p = A + (size_t)(row_base + row) * lda + k0 + half * 16;
  v16bf a;
  *(v8bf*)&a       = *(const v8bf*)(p);
  *((v8bf*)&a + 1) = *(const v8bf*)(p + 8);
  return a;
}

// B (32x16 bf16). Bt stored N-major: Bt[n][k], row stride ldb elements.
//   lanes 0-15 : col = lane,   K = k0+0..15 packed along VGPRs
//   lanes 16-31: col = lane-16, K = k0+16..31
// => 32 contiguous bytes per lane.
__device__ __forceinline__ v16bf load_b_frag(const __bf16* Bt, int ldb,
                                             int n_base, int k0, int lane) {
  const int half = lane >> 4;
  const int col  = lane & 15;
  const __bf16* p = Bt + (size_t)(n_base + col) * ldb + k0 + half * 16;
  return *(const v16bf*)p;
}

// stage a 32-row x 32-K bf16 A tile (2 KB) into LDS with 128 threads:
// one 16-byte transfer per thread.  LDS row stride = 40 elements (80 B).
__device__ __forceinline__ void stage_a_tile(__bf16* lds_dst,
                                             const __bf16* gsrc_base, int tid) {
  const int row   = tid >> 2;
  const int chunk = tid & 3;
  const __bf16* g = gsrc_base + (size_t)row * D_ + chunk * 8;
  __bf16*       l = lds_dst + row * 40 + chunk * 8;
#if HAVE_ASYNC_LDS
  __builtin_amdgcn_global_load_async_to_lds_b128((gv4i_t*)g, (lv4i_t*)l, 0, 0);
#else
  *(v8bf*)l = *(const v8bf*)g;
#endif
}

// ---- conversion kernels ----------------------------------------------------
__global__ void k_cvt(const float* __restrict__ in, __bf16* __restrict__ o, int n) {
  int i = blockIdx.x * blockDim.x + threadIdx.x;
  if (i < n) o[i] = (__bf16)in[i];
}

// w is [K][N] f32 row-major; wT is [N][K] bf16 row-major (i.e. transposed)
__global__ void k_cvtT(const float* __restrict__ w, __bf16* __restrict__ wT,
                       int K, int N) {
  int i = blockIdx.x * blockDim.x + threadIdx.x;
  if (i < K * N) {
    int nn = i / K, kk = i - nn * K;
    wT[i] = (__bf16)w[(size_t)kk * N + nn];
  }
}

// ---- kernel 1: QKV = Xb @ WqkvT  (+RoPE on Q,K; V stored transposed) -------
// block = 4 waves; shared 32-row A tile staged (async) through LDS,
// each wave owns a 32x64 output tile.
__global__ __launch_bounds__(128)
void k_qkv_rope(const __bf16* __restrict__ xb, const __bf16* __restrict__ wqkvT,
                __bf16* __restrict__ q_ws, __bf16* __restrict__ k_ws,
                __bf16* __restrict__ v_ws) {
  __shared__ __align__(32) __bf16 lds_a[2][32 * 40];
  const int tid  = threadIdx.x;
  const int lane = tid & 31;
  const int wave = tid >> 5;
  const int m0 = blockIdx.x * 32;                 // rows of X
  const int n0 = (blockIdx.y * 4 + wave) * 64;    // cols of QKV (0..3071)

  v8f acc[2][4] = {};
  stage_a_tile(&lds_a[0][0], xb + (size_t)m0 * D_, tid);
  for (int step = 0; step < D_ / 32; ++step) {
    const int k0 = step * 32;
    const __bf16* cur = &lds_a[step & 1][0];
    wait_async_done();
    __syncthreads();                               // tile `cur` visible to all
    if (step + 1 < D_ / 32) {
      stage_a_tile(&lds_a[(step + 1) & 1][0],
                   xb + (size_t)m0 * D_ + k0 + 32, tid);
      __builtin_prefetch(wqkvT + (size_t)(n0 + lane) * D_ + k0 + 32, 0, 3);
    }
    v16bf a0 = load_a_frag(cur, 40, 0,  0, lane);
    v16bf a1 = load_a_frag(cur, 40, 16, 0, lane);
    #pragma unroll
    for (int j = 0; j < 4; ++j) {
      v16bf b = load_b_frag(wqkvT, D_, n0 + 16 * j, k0, lane);
      acc[0][j] = wmma_bf16(a0, b, acc[0][j]);
      acc[1][j] = wmma_bf16(a1, b, acc[1][j]);
    }
    __syncthreads();                               // done reading `cur`
  }

  const int half   = lane >> 4;
  const int colL   = lane & 15;
  const int region = n0 >> 10;     // 0=Q, 1=K, 2=V (uniform per wave)
  const int nr0    = n0 & 1023;

  #pragma unroll
  for (int i = 0; i < 2; ++i)
  #pragma unroll
  for (int j = 0; j < 4; ++j)
  #pragma unroll
  for (int r = 0; r < 8; ++r) {
    const int m = m0 + 16 * i + r + 8 * half;
    const int b = m >> 11;           // batch
    const int l = m & (L_ - 1);      // sequence pos
    const int n = nr0 + 16 * j + colL;
    const int h = n >> 6;            // head (uniform: 64-wide tile = 1 head)
    const int d = n & 63;            // dim within head
    float v = acc[i][j][r];
    if (region < 2) {
      // RoPE: pair (2p, 2p+1) lives in adjacent lanes -> shuffle partner
      float vp  = __shfl_xor(v, 1, 32);
      // inv_freq = 10000^(-(2p)/64); ln(10000)/64 = 0.143911568...
      float ang = (float)l * __expf(-(float)(d & ~1) * 0.14391156868f);
      float s, c;
      __sincosf(ang, &s, &c);
      v = v * c + ((d & 1) ? vp * s : -vp * s);
      __bf16* dst = (region == 0) ? q_ws : k_ws;
      dst[(size_t)((b * H_ + h) * L_ + l) * DH_ + d] = (__bf16)v;
    } else {
      // V stored transposed: [b][h][dh][l]
      v_ws[(size_t)((b * H_ + h) * DH_ + d) * L_ + l] = (__bf16)v;
    }
  }
}

// ---- kernel 2: flash attention per (b,h); one wave = 16 query rows ---------
__global__ __launch_bounds__(128)
void k_attn(const __bf16* __restrict__ q_ws, const __bf16* __restrict__ k_ws,
            const __bf16* __restrict__ v_ws, __bf16* __restrict__ a_ws) {
  __shared__ __align__(32) __bf16 lds_p[4][16 * 40];  // per-wave P tile 16x32 (+pad)
  const int lane = threadIdx.x & 31;
  const int wave = threadIdx.x >> 5;
  const int bh   = blockIdx.y;                        // 0..63
  const int qrow = (blockIdx.x * 4 + wave) * 16;      // 0..2047

  const __bf16* Q = q_ws + (size_t)bh * L_ * DH_;
  const __bf16* K = k_ws + (size_t)bh * L_ * DH_;
  const __bf16* V = v_ws + (size_t)bh * DH_ * L_;

  v16bf qa0 = load_a_frag(Q, DH_, qrow, 0,  lane);
  v16bf qa1 = load_a_frag(Q, DH_, qrow, 32, lane);

  v8f acc[4] = {};
  float mrun[8], lrun[8];
  #pragma unroll
  for (int r = 0; r < 8; ++r) { mrun[r] = -1e30f; lrun[r] = 0.f; }
  const float scale = 0.125f;  // 1/sqrt(64)

  __bf16* P = &lds_p[wave][0];

  for (int kb = 0; kb < L_; kb += 32) {
    float s0[8], s1[8];
    {
      v16bf b0 = load_b_frag(K, DH_, kb, 0,  lane);
      v16bf b1 = load_b_frag(K, DH_, kb, 32, lane);
      v8f c = {};
      c = wmma_bf16(qa0, b0, c);
      c = wmma_bf16(qa1, b1, c);
      #pragma unroll
      for (int r = 0; r < 8; ++r) s0[r] = c[r] * scale;
    }
    {
      v16bf b0 = load_b_frag(K, DH_, kb + 16, 0,  lane);
      v16bf b1 = load_b_frag(K, DH_, kb + 16, 32, lane);
      v8f c = {};
      c = wmma_bf16(qa0, b0, c);
      c = wmma_bf16(qa1, b1, c);
      #pragma unroll
      for (int r = 0; r < 8; ++r) s1[r] = c[r] * scale;
    }

    float p0[8], p1[8];
    #pragma unroll
    for (int r = 0; r < 8; ++r) {
      float mx = fmaxf(s0[r], s1[r]);
      mx = fmaxf(mx, __shfl_xor(mx, 1, 32));
      mx = fmaxf(mx, __shfl_xor(mx, 2, 32));
      mx = fmaxf(mx, __shfl_xor(mx, 4, 32));
      mx = fmaxf(mx, __shfl_xor(mx, 8, 32));
      float mnew  = fmaxf(mrun[r], mx);
      float alpha = __expf(mrun[r] - mnew);
      p0[r] = __expf(s0[r] - mnew);
      p1[r] = __expf(s1[r] - mnew);
      float rs = p0[r] + p1[r];
      rs += __shfl_xor(rs, 1, 32);
      rs += __shfl_xor(rs, 2, 32);
      rs += __shfl_xor(rs, 4, 32);
      rs += __shfl_xor(rs, 8, 32);
      lrun[r] = lrun[r] * alpha + rs;
      mrun[r] = mnew;
      acc[0][r] *= alpha; acc[1][r] *= alpha;
      acc[2][r] *= alpha; acc[3][r] *= alpha;
    }

    // stage P (C layout) into LDS, reload in A layout
    const int rowadd = 8 * (lane >> 4);
    const int colL   = lane & 15;
    #pragma unroll
    for (int r = 0; r < 8; ++r) {
      P[(r + rowadd) * 40 + colL]      = (__bf16)p0[r];
      P[(r + rowadd) * 40 + 16 + colL] = (__bf16)p1[r];
    }
    asm volatile("" ::: "memory");   // keep DS store->load order (DS is in-order)
    v16bf pa = load_a_frag(P, 40, 0, 0, lane);
    asm volatile("" ::: "memory");

    #pragma unroll
    for (int jd = 0; jd < 4; ++jd) {
      v16bf vb = load_b_frag(V, L_, jd * 16, kb, lane);  // cols = dh, K = keys
      acc[jd] = wmma_bf16(pa, vb, acc[jd]);
    }
  }

  const int half = lane >> 4;
  const int colL = lane & 15;
  const int b = bh >> 4, h = bh & 15;
  #pragma unroll
  for (int jd = 0; jd < 4; ++jd)
  #pragma unroll
  for (int r = 0; r < 8; ++r) {
    const int row = qrow + r + 8 * half;
    const int dh  = jd * 16 + colL;
    float o = acc[jd][r] / lrun[r];
    a_ws[(size_t)(b * L_ + row) * D_ + h * DH_ + dh] = (__bf16)o;
  }
}

// ---- kernel 3: out = attn @ w_proj  (f32 output) ---------------------------
__global__ __launch_bounds__(128)
void k_proj(const __bf16* __restrict__ a_b, const __bf16* __restrict__ wprojT,
            float* __restrict__ out) {
  __shared__ __align__(32) __bf16 lds_a[2][32 * 40];
  const int tid  = threadIdx.x;
  const int lane = tid & 31;
  const int wave = tid >> 5;
  const int m0 = blockIdx.x * 32;
  const int n0 = (blockIdx.y * 4 + wave) * 64;

  v8f acc[2][4] = {};
  stage_a_tile(&lds_a[0][0], a_b + (size_t)m0 * D_, tid);
  for (int step = 0; step < D_ / 32; ++step) {
    const int k0 = step * 32;
    const __bf16* cur = &lds_a[step & 1][0];
    wait_async_done();
    __syncthreads();
    if (step + 1 < D_ / 32) {
      stage_a_tile(&lds_a[(step + 1) & 1][0],
                   a_b + (size_t)m0 * D_ + k0 + 32, tid);
      __builtin_prefetch(wprojT + (size_t)(n0 + lane) * D_ + k0 + 32, 0, 3);
    }
    v16bf a0 = load_a_frag(cur, 40, 0,  0, lane);
    v16bf a1 = load_a_frag(cur, 40, 16, 0, lane);
    #pragma unroll
    for (int j = 0; j < 4; ++j) {
      v16bf b = load_b_frag(wprojT, D_, n0 + 16 * j, k0, lane);
      acc[0][j] = wmma_bf16(a0, b, acc[0][j]);
      acc[1][j] = wmma_bf16(a1, b, acc[1][j]);
    }
    __syncthreads();
  }

  const int half = lane >> 4;
  const int colL = lane & 15;
  #pragma unroll
  for (int i = 0; i < 2; ++i)
  #pragma unroll
  for (int j = 0; j < 4; ++j)
  #pragma unroll
  for (int r = 0; r < 8; ++r) {
    const int m = m0 + 16 * i + r + 8 * half;
    const int n = n0 + 16 * j + colL;
    out[(size_t)m * D_ + n] = acc[i][j][r];
  }
}

// ---- launcher --------------------------------------------------------------
extern "C" void kernel_launch(void* const* d_in, const int* in_sizes, int n_in,
                              void* d_out, int out_size, void* d_ws, size_t ws_size,
                              hipStream_t stream) {
  const float* x      = (const float*)d_in[0];
  const float* w_qkv  = (const float*)d_in[1];
  const float* w_proj = (const float*)d_in[2];
  float* out = (float*)d_out;
  char*  ws  = (char*)d_ws;

  // workspace layout (bytes)                       size
  __bf16* xb     = (__bf16*)(ws + 0);            // 16,777,216  x bf16 [8192][1024]
  __bf16* wqkvT  = (__bf16*)(ws + 16777216);     //  6,291,456  w_qkv^T bf16 [3072][1024]
  __bf16* wprojT = (__bf16*)(ws + 23068672);     //  2,097,152  w_proj^T bf16 [1024][1024]
  __bf16* q_ws   = (__bf16*)(ws + 25165824);     // 16,777,216  q (roped) [64][2048][64]
  __bf16* k_ws   = (__bf16*)(ws + 41943040);     // 16,777,216  k (roped) [64][2048][64]
  __bf16* v_ws   = (__bf16*)(ws + 58720256);     // 16,777,216  v^T      [64][64][2048]
  __bf16* a_ws   = (__bf16*)(ws + 75497472);     // 16,777,216  attn out [8192][1024]
  // total: 92,274,688 bytes

  { int n = M_ * D_;     k_cvt <<<(n + 255) / 256, 256, 0, stream>>>(x, xb, n); }
  { int n = D_ * NQKV_;  k_cvtT<<<(n + 255) / 256, 256, 0, stream>>>(w_qkv, wqkvT, D_, NQKV_); }
  { int n = D_ * D_;     k_cvtT<<<(n + 255) / 256, 256, 0, stream>>>(w_proj, wprojT, D_, D_); }

  k_qkv_rope<<<dim3(M_ / 32, NQKV_ / 256), 128, 0, stream>>>(xb, wqkvT, q_ws, k_ws, v_ws);
  k_attn    <<<dim3(L_ / 64, B_ * H_),     128, 0, stream>>>(q_ws, k_ws, v_ws, a_ws);
  k_proj    <<<dim3(M_ / 32, D_ / 256),    128, 0, stream>>>(a_ws, wprojT, out);
}
